// TransformerConv1_85255100825818
// MI455X (gfx1250) — matrix operand order repaired and checked by reading, not compiled, not run on hardware
//
#include <hip/hip_runtime.h>
#include <hip/hip_bf16.h>
#include <math.h>

// ---------------------------------------------------------------------------
// TransformerConv (heads=1, dense-softmax variant) for MI455X / gfx1250.
//
// Math rewrite: softmax over the DENSE logit matrix S (alpha_i at edges, 0
// elsewhere) means P[i,j] = exp(S_ij)/Z_i with exp(0)=1 at non-edges, so
//   out[i] = (H_total + w_i * sum_{e:row=i} h[col_e]) / Z_i,
//   w_i = exp(leaky_relu(a_i)) - 1,  Z_i = N + deg_i * w_i,
//   H_total = sum_j h[j].
// Replaces the O(N^2*OUT) dense aggregation with an E-edge gather whose
// working set (h, 16MB) is L2-resident (192MB L2).
// GEMMs: v_wmma_f32_16x16x32_f16, async global->LDS double buffering.
// ---------------------------------------------------------------------------

typedef __attribute__((ext_vector_type(16))) _Float16 v16h;
typedef __attribute__((ext_vector_type(8)))  float    v8f;

#define GN    8192
#define GIN   256
#define GOUT  512
#define GE    262144

// ---------------- CDNA5 async global->LDS helpers ----------------

__device__ __forceinline__ void async_g2l_b128(uint32_t lds_off, const void* gptr) {
  // GLOBAL_LOAD_ASYNC_TO_LDS_B128 (GV mode): VDST = per-lane LDS byte offset,
  // VADDR = 64-bit global address. Tracked by ASYNCcnt.
  asm volatile("global_load_async_to_lds_b128 %0, %1, off"
               :: "v"(lds_off), "v"((unsigned long long)(uintptr_t)gptr)
               : "memory");
}

__device__ __forceinline__ void wait_async0() {
  asm volatile("s_wait_asynccnt 0x0" ::: "memory");
}

// ---------------- small utility kernels ----------------

__global__ void k_f32_to_f16(const float* __restrict__ src,
                             _Float16* __restrict__ dst, int n) {
  int i = blockIdx.x * blockDim.x + threadIdx.x;
  if (i < n) dst[i] = (_Float16)src[i];
}

__global__ void k_zero_i32(int* __restrict__ p, int n) {
  int i = blockIdx.x * blockDim.x + threadIdx.x;
  if (i < n) p[i] = 0;
}

// ---------------- WMMA GEMM: C[MxN] = A[MxK] * Bw[NxK]^T ----------------
// A row-major f16, Bw row-major f16 (weight layout: B[k][n] = Bw[n][k]).
// Block tile 128(M) x 128(N), K-step 32. 8 waves as 4(M) x 2(N); each wave
// computes 32M x 64N = 8 WMMA per K-step from 2 A-frags + 4 B-frags.
// Tiles stream memory->LDS via async b128 copies, double buffered.

#define BM 128
#define BN 128
#define BK 32
#define LDT 40              // padded LDS row stride in halfs (80B, 16B-aligned)
#define TILEH (128 * LDT)   // halfs per tile buffer

template <int ELU>
__global__ __launch_bounds__(256) void k_gemm(const _Float16* __restrict__ A,
                                              const _Float16* __restrict__ Bw,
                                              float* __restrict__ C,
                                              int M, int N, int K) {
  __shared__ __attribute__((aligned(16))) _Float16 As[2 * TILEH];
  __shared__ __attribute__((aligned(16))) _Float16 Bs[2 * TILEH];

  const int tid   = threadIdx.x;
  const int wave  = tid >> 5;
  const int lane  = tid & 31;
  const int half  = lane >> 4;   // 0: lanes 0-15, 1: lanes 16-31
  const int rm    = lane & 15;
  const int waveM = wave >> 1;   // 0..3
  const int waveN = wave & 1;    // 0..1

  const int mBlock = blockIdx.y * BM;
  const int nBlock = blockIdx.x * BN;

  v8f acc[2][4];
  const v8f vz = {0.f, 0.f, 0.f, 0.f, 0.f, 0.f, 0.f, 0.f};
#pragma unroll
  for (int s = 0; s < 2; ++s)
#pragma unroll
    for (int t = 0; t < 4; ++t) acc[s][t] = vz;

  // async copy of one K-tile pair (A: 128x32, B: 128x32 halfs) into buffer b
  auto load_tiles = [&](int k0, int b) {
#pragma unroll
    for (int i = 0; i < 2; ++i) {            // 512 chunks of 16B per matrix
      int c   = tid + i * 256;
      int row = c >> 2, seg = c & 3;         // seg: 4 x 8 halfs = 32
      uint32_t la =
          (uint32_t)(uintptr_t)&As[b * TILEH + row * LDT + seg * 8];
      async_g2l_b128(la, &A[(size_t)(mBlock + row) * K + k0 + seg * 8]);
      uint32_t lb =
          (uint32_t)(uintptr_t)&Bs[b * TILEH + row * LDT + seg * 8];
      async_g2l_b128(lb, &Bw[(size_t)(nBlock + row) * K + k0 + seg * 8]);
    }
  };

  const int nsteps = K / BK;
  load_tiles(0, 0);
  wait_async0();
  __syncthreads();

  for (int s = 0; s < nsteps; ++s) {
    const int buf = s & 1;
    if (s + 1 < nsteps) load_tiles((s + 1) * BK, buf ^ 1);

    // ---- A fragments (16x32 each, CDNA5 wave32 K-slicing) ----
    union Frag { v16h v; uint32_t u[8]; };
    Frag af[2];
#pragma unroll
    for (int sub = 0; sub < 2; ++sub)
#pragma unroll
      for (int p = 0; p < 8; ++p) {
        int kp = 2 * p + (p >= 4 ? 8 : 0) + (half ? 8 : 0);
        af[sub].u[p] = *(const uint32_t*)
            &As[buf * TILEH + (waveM * 32 + sub * 16 + rm) * LDT + kp];
      }

#pragma unroll
    for (int nt = 0; nt < 4; ++nt) {
      Frag bf;
#pragma unroll
      for (int p = 0; p < 8; ++p) {
        int kp = 2 * p + (p >= 4 ? 8 : 0) + (half ? 8 : 0);
        bf.u[p] = *(const uint32_t*)
            &Bs[buf * TILEH + (waveN * 64 + nt * 16 + rm) * LDT + kp];
      }
#pragma unroll
      for (int sub = 0; sub < 2; ++sub)
        acc[sub][nt] = __builtin_amdgcn_wmma_f32_16x16x32_f16(
            false, af[sub].v, false, bf.v, (short)0, acc[sub][nt], false, false);
    }

    wait_async0();     // next tile resident in LDS (this wave's copies)
    __syncthreads();   // all waves' copies resident, all reads of buf done
  }

  // ---- epilogue: C/D layout VGPR v -> M = v + 8*half, lane rm -> N ----
#pragma unroll
  for (int sub = 0; sub < 2; ++sub) {
    const int mG = mBlock + waveM * 32 + sub * 16;
#pragma unroll
    for (int nt = 0; nt < 4; ++nt) {
#pragma unroll
      for (int v = 0; v < 8; ++v) {
        float val = acc[sub][nt][v];
        if (ELU) val = val > 0.f ? val : (__expf(val) - 1.f);
        int mm = mG + v + half * 8;
        int nn = nBlock + waveN * 64 + nt * 16 + rm;
        C[(size_t)mm * N + nn] = val;
      }
    }
  }
}

// ---------------- per-node attention scalar: w_i = exp(lrelu(a_i)) - 1 ----

__global__ void k_alpha(const float* __restrict__ h,
                        const float* __restrict__ att_w,
                        float* __restrict__ wvec) {
  int node = blockIdx.x * (blockDim.x >> 5) + (threadIdx.x >> 5);
  int lane = threadIdx.x & 31;
  if (node >= GN) return;
  float s1 = 0.f, s2 = 0.f;
  for (int c = lane; c < GOUT; c += 32) {
    float v = h[(size_t)node * GOUT + c];
    s1 += v * att_w[c];
    s2 += v;
  }
#pragma unroll
  for (int off = 16; off > 0; off >>= 1) {
    s1 += __shfl_xor(s1, off, 32);
    s2 += __shfl_xor(s2, off, 32);
  }
  if (lane == 0) {
    float a     = s1 * s2;
    float alpha = a > 0.f ? a : 0.2f * a;
    wvec[node]  = __expf(alpha) - 1.f;
  }
}

// ---------------- H_total[c] = sum_i h[i][c] (partial + atomic) ----------

__global__ void k_colsum(const float* __restrict__ h, float* __restrict__ Htot,
                         int rowsPerChunk) {
  int c  = blockIdx.x * blockDim.x + threadIdx.x;
  int r0 = blockIdx.y * rowsPerChunk;
  if (c >= GOUT) return;
  float s = 0.f;
  for (int i = 0; i < rowsPerChunk; ++i) s += h[(size_t)(r0 + i) * GOUT + c];
  atomicAdd(&Htot[c], s);
}

// ---------------- CSR build: histogram -> scan -> scatter ----------------

__global__ void k_hist(const int* __restrict__ ei, int* __restrict__ deg) {
  int e = blockIdx.x * blockDim.x + threadIdx.x;
  if (e < GE) atomicAdd(&deg[ei[e]], 1);
}

__global__ void k_scan(const int* __restrict__ deg, int* __restrict__ rowptr) {
  // single block, 256 threads, 8192 entries -> 32 per thread
  __shared__ int part[256];
  const int t = threadIdx.x;
  const int chunk = GN / 256;
  const int base  = t * chunk;
  int s = 0;
  for (int i = 0; i < chunk; ++i) s += deg[base + i];
  part[t] = s;
  __syncthreads();
  for (int off = 1; off < 256; off <<= 1) {
    int v = (t >= off) ? part[t - off] : 0;
    __syncthreads();
    part[t] += v;
    __syncthreads();
  }
  int run = (t == 0) ? 0 : part[t - 1];
  for (int i = 0; i < chunk; ++i) {
    rowptr[base + i] = run;
    run += deg[base + i];
  }
  if (t == 255) rowptr[GN] = run;
}

__global__ void k_scatter(const int* __restrict__ ei,
                          const int* __restrict__ rowptr,
                          int* __restrict__ cursor,
                          int* __restrict__ csr_col) {
  int e = blockIdx.x * blockDim.x + threadIdx.x;
  if (e >= GE) return;
  int r = ei[e];
  int c = ei[GE + e];
  int pos = atomicAdd(&cursor[r], 1);
  csr_col[rowptr[r] + pos] = c;
}

// ------- aggregate: agg[i] = (H_total + w_i * sum_nbrs h[col]) / Z_i -----
// 512 threads per node, thread t owns channel t; h gathers hit L2.

__global__ __launch_bounds__(512) void k_agg(const float* __restrict__ h,
                                             const int* __restrict__ rowptr,
                                             const int* __restrict__ csr_col,
                                             const float* __restrict__ wvec,
                                             const float* __restrict__ Htot,
                                             _Float16* __restrict__ aggh) {
  const int node = blockIdx.x;
  const int t    = threadIdx.x;
  const int s    = rowptr[node];
  const int epos = rowptr[node + 1];
  float acc = 0.f;
  for (int e = s; e < epos; ++e) {
    int c = csr_col[e];
    acc += h[(size_t)c * GOUT + t];
  }
  const float w   = wvec[node];
  const float Z   = (float)GN + (float)(epos - s) * w;
  const float val = (Htot[t] + w * acc) / Z;
  aggh[(size_t)node * GOUT + t] = (_Float16)val;
}

// ---------------------------------------------------------------------------

extern "C" void kernel_launch(void* const* d_in, const int* in_sizes, int n_in,
                              void* d_out, int out_size, void* d_ws, size_t ws_size,
                              hipStream_t stream) {
  const float* x     = (const float*)d_in[0];  // [N, IN]
  const int*   ei    = (const int*)d_in[1];    // [2, E]
  const float* W     = (const float*)d_in[2];  // [OUT, IN]
  const float* att_w = (const float*)d_in[3];  // [1, OUT]
  const float* lin_w = (const float*)d_in[4];  // [OUT, 2*IN] = [512, 512]
  float* out = (float*)d_out;                  // [N, OUT] f32

  // ---- workspace carve-out ----
  char*  ws  = (char*)d_ws;
  size_t off = 0;
  auto alloc = [&](size_t bytes) -> void* {
    void* p = ws + off;
    off += (bytes + 255) & ~(size_t)255;
    return p;
  };
  _Float16* xh     = (_Float16*)alloc((size_t)GN * GIN * 2);    // 4 MB
  _Float16* Wh     = (_Float16*)alloc((size_t)GOUT * GIN * 2);  // 256 KB
  _Float16* lWh    = (_Float16*)alloc((size_t)GOUT * GOUT * 2); // 512 KB
  _Float16* aggh   = (_Float16*)alloc((size_t)GN * GOUT * 2);   // 8 MB
  float*    h      = (float*)alloc((size_t)GN * GOUT * 4);      // 16 MB
  float*    wvec   = (float*)alloc((size_t)GN * 4);
  float*    Htot   = (float*)alloc((size_t)GOUT * 4);
  int*      deg    = (int*)alloc((size_t)GN * 4);
  int*      cursor = (int*)alloc((size_t)GN * 4);
  int*      rowptr = (int*)alloc((size_t)(GN + 1) * 4);
  int*      csr    = (int*)alloc((size_t)GE * 4);               // 1 MB

  // ---- 1) precision conversion (f32 -> f16 for WMMA) ----
  k_f32_to_f16<<<(GN * GIN + 255) / 256, 256, 0, stream>>>(x, xh, GN * GIN);
  k_f32_to_f16<<<(GOUT * GIN + 255) / 256, 256, 0, stream>>>(W, Wh, GOUT * GIN);
  k_f32_to_f16<<<(GOUT * GOUT + 255) / 256, 256, 0, stream>>>(lin_w, lWh, GOUT * GOUT);

  // ---- 2) zero accumulators (deg, cursor, Htot) ----
  k_zero_i32<<<(GN + 255) / 256, 256, 0, stream>>>(deg, GN);
  k_zero_i32<<<(GN + 255) / 256, 256, 0, stream>>>(cursor, GN);
  k_zero_i32<<<(GOUT + 255) / 256, 256, 0, stream>>>((int*)Htot, GOUT);

  // ---- 3) h = x @ W^T  (WMMA f16 -> f32) ----
  {
    dim3 grid(GOUT / BN, GN / BM);
    k_gemm<0><<<grid, 256, 0, stream>>>(xh, Wh, h, GN, GOUT, GIN);
  }

  // ---- 4) per-node scalar w_i, global column-sum H_total ----
  k_alpha<<<(GN + 7) / 8, 256, 0, stream>>>(h, att_w, wvec);
  {
    dim3 grid((GOUT + 255) / 256, 32);
    k_colsum<<<grid, 256, 0, stream>>>(h, Htot, GN / 32);
  }

  // ---- 5) CSR build ----
  k_hist<<<(GE + 255) / 256, 256, 0, stream>>>(ei, deg);
  k_scan<<<1, 256, 0, stream>>>(deg, rowptr);
  k_scatter<<<(GE + 255) / 256, 256, 0, stream>>>(ei, rowptr, cursor, csr);

  // ---- 6) sparse aggregation + softmax normalization -> f16 ----
  k_agg<<<GN, 512, 0, stream>>>(h, rowptr, csr, wvec, Htot, aggh);

  // ---- 7) out = elu(agg @ lin_w^T)  (WMMA + ELU epilogue) ----
  {
    dim3 grid(GOUT / BN, GN / BM);
    k_gemm<1><<<grid, 256, 0, stream>>>(aggh, lWh, out, GN, GOUT, GOUT);
  }
}
